// CRNN_77335181131812
// MI455X (gfx1250) — matrix operand order repaired and checked by reading, not compiled, not run on hardware
//
#include <hip/hip_runtime.h>
#include <hip/hip_bf16.h>

// ---------------------------------------------------------------------------
// CRNN for MI455X (gfx1250, wave32, WMMA).
//   k1: conv 3x3 + ReLU + freq-maxpool           -> feat16 (T,B,64) f16
//   k2: gx = feat @ w_ih^T + b_ih (WMMA f16)     -> gx (T,B,192) f32
//   k3: sequential GRU scan, batch-partitioned over 4 WGPs (WMMA f16),
//       gx[t] streamed via async global->LDS double buffering (ASYNCcnt)
//   k4: FC (64 -> 2)                             -> out (B,2,T) f32
// ---------------------------------------------------------------------------

typedef __attribute__((ext_vector_type(16))) _Float16 v16h;
typedef __attribute__((ext_vector_type(8)))  _Float16 v8h;
typedef __attribute__((ext_vector_type(8)))  float    v8f;
typedef __attribute__((ext_vector_type(4)))  int      v4i;

union HF { v8h p[2]; v16h v; };

__device__ __forceinline__ float sigmoidf_fast(float x) {
    return 1.0f / (1.0f + __expf(-x));
}
__device__ __forceinline__ float tanhf_fast(float x) {
    float e = __expf(-2.0f * x);
    return (1.0f - e) / (1.0f + e);
}

// --- async global->LDS copy of one gate slice: 3 regions x 32B ---------------
// Same IOFFSET is applied to both the LDS and global addresses (ISA 10.x),
// so one (global base, lds base) pair + 6 immediate offsets covers the slice.
__device__ __forceinline__ void gx_async_fetch(const float* g, float* l) {
#if __has_builtin(__builtin_amdgcn_global_load_async_to_lds_b128)
    typedef __attribute__((address_space(1))) v4i gv4i_t;
    typedef __attribute__((address_space(3))) v4i lv4i_t;
    gv4i_t* gp = (gv4i_t*)g;
    lv4i_t* lp = (lv4i_t*)l;
    __builtin_amdgcn_global_load_async_to_lds_b128(gp, lp, 0,   0);
    __builtin_amdgcn_global_load_async_to_lds_b128(gp, lp, 16,  0);
    __builtin_amdgcn_global_load_async_to_lds_b128(gp, lp, 256, 0);
    __builtin_amdgcn_global_load_async_to_lds_b128(gp, lp, 272, 0);
    __builtin_amdgcn_global_load_async_to_lds_b128(gp, lp, 512, 0);
    __builtin_amdgcn_global_load_async_to_lds_b128(gp, lp, 528, 0);
#else
    unsigned l32 = (unsigned)(unsigned long long)l;        // LDS byte address
    unsigned long long g64 = (unsigned long long)g;
    asm volatile(
        "global_load_async_to_lds_b128 %0, %1, off\n\t"
        "global_load_async_to_lds_b128 %0, %1, off offset:16\n\t"
        "global_load_async_to_lds_b128 %0, %1, off offset:256\n\t"
        "global_load_async_to_lds_b128 %0, %1, off offset:272\n\t"
        "global_load_async_to_lds_b128 %0, %1, off offset:512\n\t"
        "global_load_async_to_lds_b128 %0, %1, off offset:528\n\t"
        :: "v"(l32), "v"(g64) : "memory");
#endif
}

// Wait until at most 6 async copies are outstanding: the just-issued prefetch
// for step t+1 may be in flight, everything for step t has landed in LDS.
__device__ __forceinline__ void wait_async_le6() {
#if __has_builtin(__builtin_amdgcn_s_wait_asynccnt)
    __builtin_amdgcn_s_wait_asynccnt(6);
#else
    asm volatile("s_wait_asynccnt 0x6" ::: "memory");
#endif
}

// ---------------------------------------------------------------------------
// Kernel 1: conv2d(1->64, 3x3, SAME) + ReLU + max over frequency.
// Output feat16 laid out (T, B, C) in f16 so it is directly the WMMA A matrix
// for the gx GEMM (row = t*64+b, 64 contiguous f16 per row).
// ---------------------------------------------------------------------------
__global__ __launch_bounds__(256) void k1_conv_relu_maxpool(
    const float* __restrict__ x,        // (B, F=64, T=1024)
    const float* __restrict__ cw,       // (64, 1, 3, 3)
    const float* __restrict__ cb,       // (64,)
    _Float16* __restrict__ feat16)      // (T, B, 64)
{
    const int b  = blockIdx.y;
    const int t0 = blockIdx.x * 4;          // 4 time columns per block
    const int c  = threadIdx.x & 63;
    const int ts = threadIdx.x >> 6;        // 0..3

    __shared__ float xs[64][6];             // freq rows x time cols t0-1..t0+4
    for (int i = threadIdx.x; i < 64 * 6; i += 256) {
        int f = i / 6, k = i % 6;
        int t = t0 - 1 + k;
        xs[f][k] = (t >= 0 && t < 1024) ? x[((size_t)b * 64 + f) * 1024 + t] : 0.0f;
    }
    __syncthreads();

    float w[9];
#pragma unroll
    for (int i = 0; i < 9; ++i) w[i] = cw[c * 9 + i];
    const float bias = cb[c];

    const int k = ts + 1;                   // center time column in xs
    float m = 0.0f;                         // relu outputs are >= 0
    for (int f = 0; f < 64; ++f) {
        float acc = bias;
#pragma unroll
        for (int df = -1; df <= 1; ++df) {
            int ff = f + df;
            if (ff < 0 || ff > 63) continue;        // uniform branch
            const float* wr = &w[(df + 1) * 3];
            acc += xs[ff][k - 1] * wr[0] + xs[ff][k] * wr[1] + xs[ff][k + 1] * wr[2];
        }
        m = fmaxf(m, fmaxf(acc, 0.0f));
    }
    feat16[((size_t)(t0 + ts) * 64 + b) * 64 + c] = (_Float16)m;
}

// ---------------------------------------------------------------------------
// Fragment layouts (CDNA5 ISA 7.12.2):
//   A (16x32 f16):  lane L holds row m=L%16; half j -> K = j + 8*(j>=8) + 8*(L>=16)
//   B (32x16 f16):  lane L holds col n=L%16; half j -> K = j + 8*(j>=8) + 8*(L>=16)
// B fragments are staged in LDS pre-permuted so each lane reads 16 contiguous
// halfs (two ds_load_b128).
// ---------------------------------------------------------------------------

// Kernel 2: gx GEMM.  M = T*B = 65536 rows, N = 192, K = 64.
// 8 waves per block, each wave owns one 16-row M tile, loops 12 N tiles.
__global__ __launch_bounds__(256) void k2_gx_gemm(
    const _Float16* __restrict__ feat16,   // (T*B, 64) f16
    const float* __restrict__ w_ih,        // (192, 64)
    const float* __restrict__ b_ih,        // (192,)
    float* __restrict__ gx)                // (T*B, 192) f32
{
    __shared__ _Float16 bfrag[12 * 2 * 32 * 16];   // 24 KB, fragment-packed
    const int tid = threadIdx.x;
    for (int idx = tid; idx < 12 * 2 * 32 * 16; idx += 256) {
        int j  = idx & 15;
        int L  = (idx >> 4) & 31;
        int kh = (idx >> 9) & 1;
        int nt = idx >> 10;
        int kk = kh * 32 + (j & 7) + ((j & 8) ? 16 : 0) + ((L & 16) ? 8 : 0);
        int n  = nt * 16 + (L & 15);
        bfrag[idx] = (_Float16)w_ih[n * 64 + kk];
    }
    __syncthreads();

    const int w    = tid >> 5;
    const int lane = tid & 31;
    const int hi   = (lane >> 4) & 1;
    const int mtile = blockIdx.x * 8 + w;                 // 0..4095
    const _Float16* arow = feat16 + (size_t)(mtile * 16 + (lane & 15)) * 64;

    HF a0, a1;
    a0.p[0] = *(const v8h*)(arow + 0  + 8 * hi);
    a0.p[1] = *(const v8h*)(arow + 16 + 8 * hi);
    a1.p[0] = *(const v8h*)(arow + 32 + 8 * hi);
    a1.p[1] = *(const v8h*)(arow + 48 + 8 * hi);

    for (int nt = 0; nt < 12; ++nt) {
        HF b0, b1;
        const _Float16* bp0 = &bfrag[((nt * 2 + 0) * 32 + lane) * 16];
        const _Float16* bp1 = &bfrag[((nt * 2 + 1) * 32 + lane) * 16];
        b0.p[0] = *(const v8h*)(bp0); b0.p[1] = *(const v8h*)(bp0 + 8);
        b1.p[0] = *(const v8h*)(bp1); b1.p[1] = *(const v8h*)(bp1 + 8);

        v8f c = {};
        c = __builtin_amdgcn_wmma_f32_16x16x32_f16(false, a0.v, false, b0.v,
                                                   (short)0, c, false, false);
        c = __builtin_amdgcn_wmma_f32_16x16x32_f16(false, a1.v, false, b1.v,
                                                   (short)0, c, false, false);

        int g = nt * 16 + (lane & 15);
        float bias = b_ih[g];
        float* outp = gx + (size_t)(mtile * 16 + 8 * hi) * 192 + g;
#pragma unroll
        for (int r = 0; r < 8; ++r) outp[(size_t)r * 192] = c[r] + bias;
    }
}

// ---------------------------------------------------------------------------
// Kernel 3: sequential GRU scan.  Recurrence is independent per batch row:
// 4 workgroups x 16 batch rows, 4 waves each.  Per step, per WG:
//   gh(16x192) = h16(16x64) @ w_hh^T via 12 tiles x 2 WMMAs (wave: 3 tiles),
//   then gate math (sigmoid/tanh) on 128 threads x 8 elems, h stays in LDS.
// w_hh B-fragments live in registers for the whole scan.  gx[t] is streamed
// into LDS with async global->LDS copies, double-buffered one step ahead so
// its latency hides under the WMMA phase + barrier of the previous step.
// ---------------------------------------------------------------------------
__global__ __launch_bounds__(128) void k3_gru_scan(
    const float* __restrict__ gx,          // (T*B, 192)
    const float* __restrict__ w_hh,        // (192, 64)
    const float* __restrict__ b_hh,        // (192,)
    float* __restrict__ hs)                // (T*B, 64)
{
    __shared__ _Float16 whh[12 * 2 * 32 * 16];   // 24 KB fragment-packed
    __shared__ _Float16 h16[16 * 64];            // A operand (f16 mirror of h)
    __shared__ float    h32[16 * 64];            // h in f32 for the z-blend
    __shared__ float    gh[16 * 192];            // recurrent pre-activations
    __shared__ float    gxlds[2][16 * 192];      // double-buffered gx tile (24 KB)

    const int tid = threadIdx.x;
    const int wg  = blockIdx.x;                  // batch rows wg*16 .. wg*16+15

    for (int idx = tid; idx < 12 * 2 * 32 * 16; idx += 128) {
        int j  = idx & 15;
        int L  = (idx >> 4) & 31;
        int kh = (idx >> 9) & 1;
        int nt = idx >> 10;
        int kk = kh * 32 + (j & 7) + ((j & 8) ? 16 : 0) + ((L & 16) ? 8 : 0);
        int n  = nt * 16 + (L & 15);
        whh[idx] = (_Float16)w_hh[n * 64 + kk];
    }
    for (int i = tid; i < 16 * 64; i += 128) { h16[i] = (_Float16)0.0f; h32[i] = 0.0f; }

    const int w    = tid >> 5;
    const int lane = tid & 31;
    const int hi   = (lane >> 4) & 1;

    const int brow  = tid >> 3;                  // 0..15
    const int j0    = (tid & 7) * 8;             // 0..56
    const int bglob = wg * 16 + brow;

    // kick off async fetch of gx[0] into buffer 0 (each lane owns its slice)
    gx_async_fetch(gx + ((size_t)0 * 64 + bglob) * 192 + j0,
                   &gxlds[0][brow * 192 + j0]);

    __syncthreads();

    HF bf[3][2];                                 // wave-resident w_hh fragments
#pragma unroll
    for (int i = 0; i < 3; ++i)
#pragma unroll
        for (int kh = 0; kh < 2; ++kh) {
            const _Float16* bp = &whh[(((3 * w + i) * 2 + kh) * 32 + lane) * 16];
            bf[i][kh].p[0] = *(const v8h*)(bp);
            bf[i][kh].p[1] = *(const v8h*)(bp + 8);
        }

    float bh_r[8], bh_z[8], bh_n[8];
#pragma unroll
    for (int i = 0; i < 8; ++i) {
        bh_r[i] = b_hh[j0 + i];
        bh_z[i] = b_hh[64 + j0 + i];
        bh_n[i] = b_hh[128 + j0 + i];
    }

    for (int t = 0; t < 1024; ++t) {
        // --- async prefetch of next step's gx tile into the other buffer ---
        // (t==1023 prefetches one row past gx; that lands in the hs region of
        //  the same workspace allocation -- valid memory, value unused)
        gx_async_fetch(gx + ((size_t)(t + 1) * 64 + bglob) * 192 + j0,
                       &gxlds[(t + 1) & 1][brow * 192 + j0]);

        // --- recurrent GEMM: A fragments from LDS h16 ---
        const _Float16* ar = &h16[(lane & 15) * 64];
        HF a0, a1;
        a0.p[0] = *(const v8h*)(ar + 0  + 8 * hi);
        a0.p[1] = *(const v8h*)(ar + 16 + 8 * hi);
        a1.p[0] = *(const v8h*)(ar + 32 + 8 * hi);
        a1.p[1] = *(const v8h*)(ar + 48 + 8 * hi);
#pragma unroll
        for (int i = 0; i < 3; ++i) {
            v8f c = {};
            c = __builtin_amdgcn_wmma_f32_16x16x32_f16(false, a0.v, false, bf[i][0].v,
                                                       (short)0, c, false, false);
            c = __builtin_amdgcn_wmma_f32_16x16x32_f16(false, a1.v, false, bf[i][1].v,
                                                       (short)0, c, false, false);
            int nt = 3 * w + i;
            float* gp = &gh[(8 * hi) * 192 + nt * 16 + (lane & 15)];
#pragma unroll
            for (int r = 0; r < 8; ++r) gp[r * 192] = c[r];
        }
        __syncthreads();

        // --- gate math (each thread owns 8 hidden units of one batch row) ---
        wait_async_le6();        // current gx buffer resident (in-order per wave)
        const float* gxp = &gxlds[t & 1][brow * 192 + j0];
        float hnew[8];
#pragma unroll
        for (int i = 0; i < 8; ++i) {
            int j = j0 + i;
            float xr = gxp[i], xz = gxp[64 + i], xn = gxp[128 + i];
            float hr = gh[brow * 192 + j]       + bh_r[i];
            float hz = gh[brow * 192 + 64 + j]  + bh_z[i];
            float hn = gh[brow * 192 + 128 + j] + bh_n[i];
            float r = sigmoidf_fast(xr + hr);
            float z = sigmoidf_fast(xz + hz);
            float n = tanhf_fast(xn + r * hn);
            float h = h32[brow * 64 + j];
            hnew[i] = (1.0f - z) * n + z * h;
        }
        float* hsp = hs + ((size_t)t * 64 + bglob) * 64 + j0;
#pragma unroll
        for (int i = 0; i < 8; ++i) {
            h32[brow * 64 + j0 + i] = hnew[i];
            h16[brow * 64 + j0 + i] = (_Float16)hnew[i];
            hsp[i] = hnew[i];
        }
        __syncthreads();
    }
}

// ---------------------------------------------------------------------------
// Kernel 4: FC (64 -> 2), out layout (B, 2, T).
// ---------------------------------------------------------------------------
__global__ __launch_bounds__(256) void k4_fc(
    const float* __restrict__ hs,          // (T*B, 64)
    const float* __restrict__ fc_w,        // (2, 64)
    const float* __restrict__ fc_b,        // (2,)
    float* __restrict__ out)               // (B, 2, T)
{
    int id = blockIdx.x * 256 + threadIdx.x;   // 0..65535 = t*64 + b
    int t = id >> 6, b = id & 63;
    const float* hp = hs + (size_t)id * 64;
    float a0 = fc_b[0], a1 = fc_b[1];
#pragma unroll
    for (int h = 0; h < 64; h += 4) {
        float4 hv = *(const float4*)(hp + h);
        float4 w0 = *(const float4*)(fc_w + h);
        float4 w1 = *(const float4*)(fc_w + 64 + h);
        a0 += hv.x * w0.x + hv.y * w0.y + hv.z * w0.z + hv.w * w0.w;
        a1 += hv.x * w1.x + hv.y * w1.y + hv.z * w1.z + hv.w * w1.w;
    }
    out[(size_t)b * 2048 + t]        = a0;
    out[(size_t)b * 2048 + 1024 + t] = a1;
}

// ---------------------------------------------------------------------------
extern "C" void kernel_launch(void* const* d_in, const int* in_sizes, int n_in,
                              void* d_out, int out_size, void* d_ws, size_t ws_size,
                              hipStream_t stream) {
    const float* x      = (const float*)d_in[0];
    const float* conv_w = (const float*)d_in[1];
    const float* conv_b = (const float*)d_in[2];
    const float* w_ih   = (const float*)d_in[3];
    const float* w_hh   = (const float*)d_in[4];
    const float* b_ih   = (const float*)d_in[5];
    const float* b_hh   = (const float*)d_in[6];
    const float* fc_w   = (const float*)d_in[7];
    const float* fc_b   = (const float*)d_in[8];
    float* out = (float*)d_out;

    char* ws = (char*)d_ws;
    _Float16* feat16 = (_Float16*)ws;                      //  8 MB  (T,B,64) f16
    float*    gxbuf  = (float*)(ws + (size_t)(8u  << 20)); // 48 MB  (T,B,192) f32
    float*    hsbuf  = (float*)(ws + (size_t)(56u << 20)); // 16 MB  (T,B,64)  f32

    k1_conv_relu_maxpool<<<dim3(256, 64), 256, 0, stream>>>(x, conv_w, conv_b, feat16);
    k2_gx_gemm<<<512, 256, 0, stream>>>(feat16, w_ih, b_ih, gxbuf);
    k3_gru_scan<<<4, 128, 0, stream>>>(gxbuf, w_hh, b_hh, hsbuf);
    k4_fc<<<256, 256, 0, stream>>>(hsbuf, fc_w, fc_b, out);
}
